// SPELTnet_61194694033667
// MI455X (gfx1250) — compile-verified, matrix operand order checked by reading, not run on hardware
//
#include <hip/hip_runtime.h>
#include <hip/hip_bf16.h>
#include <math.h>

// ---------------------------------------------------------------------------
// MI455X (gfx1250) implementation.
//   N = 100000 rows, F_in = 256, D = 128, E = 64, K(topk) = 5.
//   Main GEMM: v_wmma_f32_16x16x32_bf16 (f32 accumulate), W staged in LDS.
// ---------------------------------------------------------------------------

#define FIN   256
#define DPROJ 128
#define EEDGE 64
#define TOPK  5
#define SPAD  129   // padded LDS stride for [64 x 128] accumulators (bank = e + d)
#define WPAD  264   // padded bf16 stride for W rows in LDS (528B -> 4-bank shift/row)

typedef __attribute__((ext_vector_type(16))) __bf16 v16bf;
typedef __attribute__((ext_vector_type(8)))  float  v8f;

union BFrag {
    v16bf  v;
    __bf16 h[16];
    uint4  u[2];
};

// ---------------------------------------------------------------------------
// Kernel 0: zero global scatter buffers, W (f32) -> bf16, normalize centers.
// ---------------------------------------------------------------------------
__global__ void __launch_bounds__(256)
k_init(const float* __restrict__ W, const float* __restrict__ centers,
       __bf16* __restrict__ Wbf, float* __restrict__ cn,
       float* __restrict__ SDe /* S[3*64*128] then De[3*64], contiguous */)
{
    const int gid    = blockIdx.x * 256 + threadIdx.x;
    const int stride = gridDim.x * 256;

    for (int i = gid; i < 3 * EEDGE * DPROJ + 3 * EEDGE; i += stride)
        SDe[i] = 0.0f;

    for (int i = gid; i < DPROJ * FIN; i += stride)
        Wbf[i] = (__bf16)W[i];

    if (gid < EEDGE) {
        float ss = 0.0f;
        for (int d = 0; d < DPROJ; ++d) {
            float c = centers[gid * DPROJ + d];
            ss += c * c;
        }
        float nn  = sqrtf(ss);
        float inv = (nn > 0.0f) ? 1.0f / nn : 1.0f;
        for (int d = 0; d < DPROJ; ++d)
            cn[gid * DPROJ + d] = centers[gid * DPROJ + d] * inv;
    }
}

// ---------------------------------------------------------------------------
// Kernel 1: x_proj = tanh(x @ W^T + b) -> d_out (overwritten in place later).
// 256 threads = 8 waves; each wave owns a 16-row M-tile, 8 f32 accumulators
// cover all 128 output columns, K = 256 in 8 WMMA steps of 32.
// W (bf16) is staged once per WG into LDS with a 528B row pitch so the 16
// rows of a B-fragment hit disjoint bank groups; B-fragments come from
// ds_load_b128 while the x-stream owns the VMEM path.
// A-fragment layout (16-bit 16x32): lane<16 -> K {0..7,16..23},
// lane>=16 -> K {8..15,24..31}.
// ---------------------------------------------------------------------------
__global__ void __launch_bounds__(256)
k_proj(const float* __restrict__ A, const float* __restrict__ P,
       const float* __restrict__ Ng, const __bf16* __restrict__ Wbf,
       const float* __restrict__ bias, float* __restrict__ out, int N)
{
    __shared__ __bf16 Wl[DPROJ * WPAD];   // 128 * 264 * 2B = 67584 B

    const int tid = threadIdx.x;
    {
        // cooperative stage: 2 threads per W row, 256B (16 x uint4) each
        const int row  = tid >> 1;
        const int half = tid & 1;
        const uint4* src =
            reinterpret_cast<const uint4*>(Wbf + (size_t)row * FIN + half * 128);
        uint4* dst = reinterpret_cast<uint4*>(&Wl[row * WPAD + half * 128]);
        #pragma unroll
        for (int i = 0; i < 16; ++i) dst[i] = src[i];
    }
    __syncthreads();

    const int lane  = tid & 31;
    const int wave  = tid >> 5;
    const int mat   = blockIdx.y;
    const float* x  = (mat == 0) ? A : ((mat == 1) ? P : Ng);
    float* xp       = out + (size_t)mat * N * DPROJ;

    const int m0 = (blockIdx.x * 8 + wave) * 16;
    if (m0 >= N) return;                  // wave-uniform; no barriers below

    const int lmod  = lane & 15;
    const int lhalf = lane >> 4;
    int lrow = m0 + lmod;
    if (lrow >= N) lrow = N - 1;          // clamp loads; stores are guarded
    const float* xrow = x + (size_t)lrow * FIN;

    v8f acc[8] = {};

    for (int kc = 0; kc < 8; ++kc) {
        const int koff = kc * 32 + lhalf * 8;
        if (kc < 7) __builtin_prefetch(xrow + koff + 32, 0, 1);

        const float4* ap = reinterpret_cast<const float4*>(xrow + koff);
        float4 f0 = ap[0], f1 = ap[1];    // K = koff .. koff+7
        float4 f2 = ap[4], f3 = ap[5];    // K = koff+16 .. koff+23

        BFrag afr;
        afr.h[0]  = (__bf16)f0.x; afr.h[1]  = (__bf16)f0.y;
        afr.h[2]  = (__bf16)f0.z; afr.h[3]  = (__bf16)f0.w;
        afr.h[4]  = (__bf16)f1.x; afr.h[5]  = (__bf16)f1.y;
        afr.h[6]  = (__bf16)f1.z; afr.h[7]  = (__bf16)f1.w;
        afr.h[8]  = (__bf16)f2.x; afr.h[9]  = (__bf16)f2.y;
        afr.h[10] = (__bf16)f2.z; afr.h[11] = (__bf16)f2.w;
        afr.h[12] = (__bf16)f3.x; afr.h[13] = (__bf16)f3.y;
        afr.h[14] = (__bf16)f3.z; afr.h[15] = (__bf16)f3.w;

        #pragma unroll
        for (int nt = 0; nt < 8; ++nt) {
            const uint4* bp = reinterpret_cast<const uint4*>(
                &Wl[(nt * 16 + lmod) * WPAD + koff]);
            BFrag bfr;
            bfr.u[0] = bp[0];             // 8 bf16: K = koff .. koff+7
            bfr.u[1] = bp[2];             // 8 bf16: K = koff+16 .. koff+23
            acc[nt] = __builtin_amdgcn_wmma_f32_16x16x32_bf16(
                false, afr.v, false, bfr.v, (short)0, acc[nt], false, false);
        }
    }

    // C/D layout: lane<16 -> rows m0+0..7, lane>=16 -> rows m0+8..15,
    // column = nt*16 + (lane&15).
    const int r0 = lhalf * 8;
    #pragma unroll
    for (int nt = 0; nt < 8; ++nt) {
        const int col  = nt * 16 + lmod;
        const float bb = bias[col];
        #pragma unroll
        for (int r = 0; r < 8; ++r) {
            const int row = m0 + r0 + r;
            if (row < N)
                xp[(size_t)row * DPROJ + col] = tanhf(acc[nt][r] + bb);
        }
    }
}

// ---------------------------------------------------------------------------
// Kernel 2: per-row cosine sims vs normalized centers, streaming top-5,
// dv^-1/2, LDS-staged scatter of S[e][d] += val*dv*xp[d] and De[e] += val.
// ---------------------------------------------------------------------------
__global__ void __launch_bounds__(256)
k_edges(const float* __restrict__ out, const float* __restrict__ cn,
        float* __restrict__ vals, int* __restrict__ idxs,
        float* __restrict__ dvis, float* __restrict__ Sg,
        float* __restrict__ Deg, int N)
{
    __shared__ float S_l[EEDGE * SPAD];   // 33024 B, bank = e + d
    __shared__ float De_l[EEDGE];
    const int tid = threadIdx.x;

    for (int i = tid; i < EEDGE * SPAD; i += 256) S_l[i] = 0.0f;
    if (tid < EEDGE) De_l[tid] = 0.0f;
    __syncthreads();

    const int mat = blockIdx.y;
    const int row = blockIdx.x * 256 + tid;

    if (row < N) {
        const float* xp = out + ((size_t)mat * N + row) * DPROJ;
        const float4* x4 = reinterpret_cast<const float4*>(xp);
        float4 xv[32];
        float ss = 0.0f;
        #pragma unroll
        for (int q = 0; q < 32; ++q) {
            xv[q] = x4[q];
            ss += xv[q].x * xv[q].x + xv[q].y * xv[q].y +
                  xv[q].z * xv[q].z + xv[q].w * xv[q].w;
        }
        float nn  = sqrtf(ss);
        float inv = (nn > 0.0f) ? 1.0f / nn : 1.0f;

        float tv[TOPK];
        int   ti[TOPK];
        #pragma unroll
        for (int k = 0; k < TOPK; ++k) { tv[k] = -2.0f; ti[k] = 0; }

        for (int e = 0; e < EEDGE; ++e) {
            const float4* c4 = reinterpret_cast<const float4*>(cn + e * DPROJ);
            float s = 0.0f;
            #pragma unroll
            for (int q = 0; q < 32; ++q) {
                float4 c = c4[q];
                s += xv[q].x * c.x + xv[q].y * c.y +
                     xv[q].z * c.z + xv[q].w * c.w;
            }
            s = fminf(1.0f, fmaxf(-1.0f, s * inv));
            // stable descending insert (strict > keeps earlier index on ties,
            // matching lax.top_k)
            float cv = s; int ci = e;
            #pragma unroll
            for (int k = 0; k < TOPK; ++k) {
                bool sw   = cv > tv[k];
                float t0  = sw ? tv[k] : cv;
                int   t1  = sw ? ti[k] : ci;
                tv[k]     = sw ? cv : tv[k];
                ti[k]     = sw ? ci : ti[k];
                cv = t0; ci = t1;
            }
        }

        float dsum = tv[0] + tv[1] + tv[2] + tv[3] + tv[4];
        float dv   = 1.0f / (sqrtf(dsum) + 1e-6f);

        const size_t rbase = (size_t)mat * N + row;
        #pragma unroll
        for (int k = 0; k < TOPK; ++k) {
            vals[rbase * TOPK + k] = tv[k];
            idxs[rbase * TOPK + k] = ti[k];
        }
        dvis[rbase] = dv;

        float sc[TOPK];
        #pragma unroll
        for (int k = 0; k < TOPK; ++k) {
            sc[k] = tv[k] * dv;
            atomicAdd(&De_l[ti[k]], tv[k]);
        }
        #pragma unroll
        for (int q = 0; q < 32; ++q) {
            float4 xq = xv[q];
            #pragma unroll
            for (int k = 0; k < TOPK; ++k) {
                float* p = &S_l[ti[k] * SPAD + q * 4];
                atomicAdd(p + 0, sc[k] * xq.x);
                atomicAdd(p + 1, sc[k] * xq.y);
                atomicAdd(p + 2, sc[k] * xq.z);
                atomicAdd(p + 3, sc[k] * xq.w);
            }
        }
    }
    __syncthreads();

    for (int i = tid; i < EEDGE * DPROJ; i += 256) {
        int e = i >> 7, d = i & 127;
        atomicAdd(&Sg[(size_t)mat * EEDGE * DPROJ + i], S_l[e * SPAD + d]);
    }
    if (tid < EEDGE) atomicAdd(&Deg[mat * EEDGE + tid], De_l[tid]);
}

// ---------------------------------------------------------------------------
// Kernel 3: S[mat][e][:] *= 1/(De[mat][e] + eps)   (step3, in place)
// ---------------------------------------------------------------------------
__global__ void __launch_bounds__(256)
k_scale(float* __restrict__ S, const float* __restrict__ De)
{
    const int i = blockIdx.x * 256 + threadIdx.x;
    if (i < 3 * EEDGE * DPROJ) {
        int mat = i >> 13;
        int e   = (i >> 7) & 63;
        S[i] *= 1.0f / (De[mat * EEDGE + e] + 1e-6f);
    }
}

// ---------------------------------------------------------------------------
// Kernel 4 (fused): for each row, for all 3 matrices
//   out = (2-alpha)*x_proj + alpha*dv*sum_k val_k*step3[idx_k]
// and directly dis = ||a-n+eps|| - ||a-p+eps||  (saves a 154MB re-read pass).
// All 3 step3 blocks staged in LDS (3*64*129*4 = 99072 B < 320KB/WG).
// ---------------------------------------------------------------------------
__global__ void __launch_bounds__(256)
k_diffuse(float* __restrict__ out, const float* __restrict__ S,
          const float* __restrict__ vals, const int* __restrict__ idxs,
          const float* __restrict__ dvis, const float* __restrict__ alphaPtr,
          float* __restrict__ dis, int N)
{
    __shared__ float S_l[3 * EEDGE * SPAD];
    const int tid = threadIdx.x;
    for (int i = tid; i < 3 * EEDGE * DPROJ; i += 256) {
        int m = i >> 13;
        int r = i & 8191;
        int e = r >> 7, d = r & 127;
        S_l[m * (EEDGE * SPAD) + e * SPAD + d] = S[i];
    }
    __syncthreads();

    const int row = blockIdx.x * 256 + tid;
    if (row >= N) return;

    const float alpha = alphaPtr[0];
    const float ca    = 2.0f - alpha;

    float c[3][TOPK];
    int   off[3][TOPK];
    float cb[3];
    #pragma unroll
    for (int m = 0; m < 3; ++m) {
        const size_t rbase = (size_t)m * N + row;
        cb[m] = alpha * dvis[rbase];
        #pragma unroll
        for (int k = 0; k < TOPK; ++k) {
            c[m][k]   = vals[rbase * TOPK + k];
            off[m][k] = m * (EEDGE * SPAD) + idxs[rbase * TOPK + k] * SPAD;
        }
    }

    float* ra = out + (size_t)row * DPROJ;
    float* rp = out + ((size_t)N + row) * DPROJ;
    float* rn = out + ((size_t)2 * N + row) * DPROJ;

    float sn = 0.0f, spd = 0.0f;

    #pragma unroll
    for (int q = 0; q < 4; ++q) {
        float av[32];
        #pragma unroll
        for (int j = 0; j < 32; ++j) {
            const int d = q * 32 + j;
            float st = c[0][0] * S_l[off[0][0] + d] + c[0][1] * S_l[off[0][1] + d] +
                       c[0][2] * S_l[off[0][2] + d] + c[0][3] * S_l[off[0][3] + d] +
                       c[0][4] * S_l[off[0][4] + d];
            av[j] = ca * ra[d] + cb[0] * st;
            ra[d] = av[j];
        }
        #pragma unroll
        for (int j = 0; j < 32; ++j) {
            const int d = q * 32 + j;
            float st = c[1][0] * S_l[off[1][0] + d] + c[1][1] * S_l[off[1][1] + d] +
                       c[1][2] * S_l[off[1][2] + d] + c[1][3] * S_l[off[1][3] + d] +
                       c[1][4] * S_l[off[1][4] + d];
            float pv = ca * rp[d] + cb[1] * st;
            rp[d] = pv;
            float dp = av[j] - pv + 1e-6f;
            spd += dp * dp;
        }
        #pragma unroll
        for (int j = 0; j < 32; ++j) {
            const int d = q * 32 + j;
            float st = c[2][0] * S_l[off[2][0] + d] + c[2][1] * S_l[off[2][1] + d] +
                       c[2][2] * S_l[off[2][2] + d] + c[2][3] * S_l[off[2][3] + d] +
                       c[2][4] * S_l[off[2][4] + d];
            float nv = ca * rn[d] + cb[2] * st;
            rn[d] = nv;
            float dn = av[j] - nv + 1e-6f;
            sn += dn * dn;
        }
    }
    dis[row] = sqrtf(sn) - sqrtf(spd);
}

// ---------------------------------------------------------------------------
extern "C" void kernel_launch(void* const* d_in, const int* in_sizes, int n_in,
                              void* d_out, int out_size, void* d_ws,
                              size_t ws_size, hipStream_t stream)
{
    (void)n_in; (void)out_size; (void)ws_size;

    const float* A       = (const float*)d_in[0];
    const float* P       = (const float*)d_in[1];
    const float* Ng      = (const float*)d_in[2];
    const float* W       = (const float*)d_in[3];
    const float* bias    = (const float*)d_in[4];
    const float* centers = (const float*)d_in[5];
    const float* alpha   = (const float*)d_in[6];

    const int N = in_sizes[0] / FIN;

    // workspace carving (bytes)
    char* ws = (char*)d_ws;
    __bf16* Wbf  = (__bf16*)(ws);                      // 128*256*2  = 65536
    float*  cn   = (float*)(ws + 65536);               // 64*128*4   = 32768
    float*  S    = (float*)(ws + 98304);               // 3*64*128*4 = 98304
    float*  De   = (float*)(ws + 196608);              // 3*64*4     = 768 (contig w/ S)
    float*  vals = (float*)(ws + 197376);              // 3*N*5*4
    int*    idxs = (int*)  (ws + 197376 + (size_t)3 * N * TOPK * 4);
    float*  dvis = (float*)(ws + 197376 + (size_t)6 * N * TOPK * 4);

    float* out = (float*)d_out;
    float* dis = out + (size_t)3 * N * DPROJ;

    k_init<<<64, 256, 0, stream>>>(W, centers, Wbf, cn, S);

    dim3 gProj((N + 127) / 128, 3);
    k_proj<<<gProj, 256, 0, stream>>>(A, P, Ng, Wbf, bias, out, N);

    dim3 gRow((N + 255) / 256, 3);
    k_edges<<<gRow, 256, 0, stream>>>(out, cn, vals, idxs, dvis, S, De, N);

    k_scale<<<96, 256, 0, stream>>>(S, De);

    k_diffuse<<<(N + 255) / 256, 256, 0, stream>>>(out, S, vals, idxs, dvis,
                                                   alpha, dis, N);
}
